// Attention_6373731467373
// MI455X (gfx1250) — matrix operand order repaired and checked by reading, not compiled, not run on hardware
//
#include <hip/hip_runtime.h>
#include <hip/hip_bf16.h>

typedef __attribute__((ext_vector_type(16))) __bf16 v16bf;
typedef __attribute__((ext_vector_type(8)))  __bf16 v8bf;
typedef __attribute__((ext_vector_type(8)))  float  v8f;

#define DIMC  768
#define NH    12
#define HD    64
#define SEQ   1024
#define BATCH 8
#define QKVN  2304            // 3*DIMC
#define MTOT  (BATCH*SEQ)     // 8192

union Frag16 { v16bf v; v8bf h[2]; };

// Load a 16-element bf16 fragment as two aligned 16-byte LDS reads.
__device__ inline v16bf ld_frag(const __bf16* lo, const __bf16* hi) {
    Frag16 f;
    f.h[0] = *(const v8bf*)lo;
    f.h[1] = *(const v8bf*)hi;
    return f.v;
}

__device__ inline v8f wmma_bf16(v16bf a, v16bf b, v8f c) {
    return __builtin_amdgcn_wmma_f32_16x16x32_bf16(false, a, false, b,
                                                   (short)0, c, false, false);
}

// ---- CDNA5 async global->LDS copy (ASYNCcnt-tracked) ----------------------
typedef int v4i __attribute__((vector_size(16)));
typedef __attribute__((address_space(1))) v4i* gp_b128;
typedef __attribute__((address_space(3))) v4i* lp_b128;

__device__ inline void async_copy_b128(const void* gsrc, void* ldst) {
#if __has_builtin(__builtin_amdgcn_global_load_async_to_lds_b128)
    __builtin_amdgcn_global_load_async_to_lds_b128(
        (gp_b128)gsrc, (lp_b128)ldst, 0, 0);
#else
    unsigned           lds = (unsigned)(uintptr_t)ldst;   // low 32b = LDS offset
    unsigned long long ga  = (unsigned long long)(uintptr_t)gsrc;
    asm volatile("global_load_async_to_lds_b128 %0, %1, off"
                 :: "v"(lds), "v"(ga) : "memory");
#endif
}

__device__ inline void wait_async0() {
#if __has_builtin(__builtin_amdgcn_s_wait_asynccnt)
    __builtin_amdgcn_s_wait_asynccnt(0);
#else
    asm volatile("s_wait_asynccnt 0x0" ::: "memory");
#endif
}

// ---------------------------------------------------------------------------
// Kernel 1: qkv = x @ w_qkv.T  (fp32 in -> bf16 out, split into Q,K,V)
// Q: [B,H,N,hd] (pre-scaled by hd^-0.5), K: [B,H,N,hd], V transposed: [B,H,hd,N]
// ---------------------------------------------------------------------------
__global__ __launch_bounds__(128)
void qkv_gemm_kernel(const float* __restrict__ X, const float* __restrict__ W,
                     __bf16* __restrict__ Q, __bf16* __restrict__ K,
                     __bf16* __restrict__ V)
{
    __shared__ __align__(16) __bf16 sA[64 * 32];
    __shared__ __align__(16) __bf16 sB[64 * 32];

    const int tid  = threadIdx.x;
    const int lane = tid & 31;
    const int wave = tid >> 5;
    const int lg   = lane >> 4;
    const int lm   = lane & 15;
    const int m0   = blockIdx.x * 64;
    const int n0   = blockIdx.y * 64;

    v8f acc[4] = {};

    for (int kb = 0; kb < DIMC; kb += 32) {
        __syncthreads();
        // Batch all global loads first (keep 8 b128 loads in flight), then cvt.
        float4 gx[4], gw[4];
        #pragma unroll
        for (int it = 0; it < 4; ++it) {
            int idx = tid + it * 128;              // 512 float4 per tile
            int row = idx >> 3, c4 = idx & 7;
            gx[it] = *(const float4*)(X + (size_t)(m0 + row) * DIMC + kb + c4 * 4);
            gw[it] = *(const float4*)(W + (size_t)(n0 + row) * DIMC + kb + c4 * 4);
        }
        if (kb + 32 < DIMC) {   // prefetch next K-chunk into cache
            __builtin_prefetch(X + (size_t)(m0 + (tid >> 3)) * DIMC + kb + 32, 0, 1);
            __builtin_prefetch(W + (size_t)(n0 + (tid >> 3)) * DIMC + kb + 32, 0, 1);
        }
        #pragma unroll
        for (int it = 0; it < 4; ++it) {
            int idx = tid + it * 128;
            int row = idx >> 3, c4 = idx & 7;
            __bf16* da = &sA[row * 32 + c4 * 4];
            da[0] = (__bf16)gx[it].x; da[1] = (__bf16)gx[it].y;
            da[2] = (__bf16)gx[it].z; da[3] = (__bf16)gx[it].w;
            __bf16* db = &sB[row * 32 + c4 * 4];
            db[0] = (__bf16)gw[it].x; db[1] = (__bf16)gw[it].y;
            db[2] = (__bf16)gw[it].z; db[3] = (__bf16)gw[it].w;
        }
        __syncthreads();

        int arow = wave * 16 + lm;
        v16bf a = ld_frag(&sA[arow * 32 + lg * 8], &sA[arow * 32 + 16 + lg * 8]);
        #pragma unroll
        for (int t = 0; t < 4; ++t) {
            int brow = t * 16 + lm;
            v16bf b = ld_frag(&sB[brow * 32 + lg * 8], &sB[brow * 32 + 16 + lg * 8]);
            acc[t] = wmma_bf16(a, b, acc[t]);
        }
    }

    #pragma unroll
    for (int t = 0; t < 4; ++t) {
        int ncol = n0 + t * 16 + lm;
        int sec  = ncol / DIMC;        // 0=Q 1=K 2=V
        int jj   = ncol % DIMC;
        int h = jj >> 6, d = jj & 63;
        #pragma unroll
        for (int r = 0; r < 8; ++r) {
            int mrow = m0 + wave * 16 + lg * 8 + r;
            int b = mrow >> 10, pos = mrow & 1023;
            float val = acc[t][r];
            if (sec == 0)
                Q[(((size_t)(b * NH + h) * SEQ + pos) << 6) + d] = (__bf16)(val * 0.125f);
            else if (sec == 1)
                K[(((size_t)(b * NH + h) * SEQ + pos) << 6) + d] = (__bf16)val;
            else
                V[((size_t)(b * NH + h) * HD + d) * SEQ + pos] = (__bf16)val;
        }
    }
}

// ---------------------------------------------------------------------------
// Kernel 2: flash attention per (b,h), 64 query rows per block, 4 waves.
// S^T = K * Q^T so post-softmax P lands directly in the A-fragment layout.
// Double-buffered LDS; K/V^T/Q staged with async global->LDS copies.
// ---------------------------------------------------------------------------
__global__ __launch_bounds__(128)
void flash_attn_kernel(const __bf16* __restrict__ Q, const __bf16* __restrict__ K,
                       const __bf16* __restrict__ V,
                       const float* __restrict__ bias_table,
                       const unsigned char* __restrict__ kmask,
                       __bf16* __restrict__ O)
{
    __shared__ __align__(16) __bf16 sQ[64 * 64];        // [qrow][dim]
    __shared__ __align__(16) __bf16 sK[2][64 * 64];     // [key][dim]
    __shared__ __align__(16) __bf16 sV[2][64 * 64];     // [dim][key] (V^T)
    __shared__ float sBias[2][128];
    __shared__ float sMask[2][64];

    const int tid  = threadIdx.x;
    const int lane = tid & 31;
    const int wave = tid >> 5;
    const int lg   = lane >> 4;
    const int lm   = lane & 15;
    const int qblk = blockIdx.x;          // 0..15
    const int bh   = blockIdx.y;          // 0..95
    const int b    = bh / NH, h = bh % NH;
    const int qbase = qblk * 64;

    auto issue_kv = [&](int j, int buf) {
        const __bf16* gk = K + ((size_t)bh * SEQ + j * 64) * HD;
        #pragma unroll
        for (int it = 0; it < 4; ++it) {
            int idx = tid + it * 128;                 // 512 x b128
            async_copy_b128(gk + idx * 8, &sK[buf][idx * 8]);
        }
        #pragma unroll
        for (int it = 0; it < 4; ++it) {
            int idx = tid + it * 128;
            int row = idx >> 3, c = idx & 7;
            async_copy_b128(V + ((size_t)bh * HD + row) * SEQ + j * 64 + c * 8,
                            &sV[buf][idx * 8]);
        }
    };
    auto stage_bm = [&](int j, int buf) {
        if (tid < 127) {
            int diff = j * 64 - qbase - 63 + tid;     // in [-1023, 1023]
            sBias[buf][tid] = bias_table[(size_t)(diff + SEQ - 1) * NH + h];
        }
        if (tid < 64)
            sMask[buf][tid] = kmask[b * SEQ + j * 64 + tid] ? 0.0f : -3.0e38f;
    };

    {   // async-stage Q tile + first K/V block + first bias/mask window
        const __bf16* gq = Q + ((size_t)bh * SEQ + qbase) * HD;
        #pragma unroll
        for (int it = 0; it < 4; ++it) {
            int idx = tid + it * 128;
            async_copy_b128(gq + idx * 8, &sQ[idx * 8]);
        }
        issue_kv(0, 0);
        stage_bm(0, 0);
    }

    v16bf qb[2];
    float m_i = -3.0e38f, l_i = 0.0f;
    v8f o[4] = {};
    const int qrl = wave * 16 + lm;

    for (int j = 0; j < SEQ / 64; ++j) {
        const int cur = j & 1;
        wait_async0();
        __syncthreads();
        if (j == 0) {   // Q^T B-fragments: lane = qrow, K = dims (registers forever)
            int row = wave * 16 + lm;
            qb[0] = ld_frag(&sQ[row * 64 +      lg * 8], &sQ[row * 64 + 16 + lg * 8]);
            qb[1] = ld_frag(&sQ[row * 64 + 32 + lg * 8], &sQ[row * 64 + 48 + lg * 8]);
        }
        if (j + 1 < SEQ / 64) {   // overlap next block's HBM traffic with compute
            issue_kv(j + 1, cur ^ 1);
            stage_bm(j + 1, cur ^ 1);
        }

        // S^T = K * Q^T : 4 key tiles, two K=32 chunks each
        v8f st[4];
        #pragma unroll
        for (int t = 0; t < 4; ++t) {
            int krow = t * 16 + lm;
            v16bf a0 = ld_frag(&sK[cur][krow * 64 +      lg * 8], &sK[cur][krow * 64 + 16 + lg * 8]);
            v16bf a1 = ld_frag(&sK[cur][krow * 64 + 32 + lg * 8], &sK[cur][krow * 64 + 48 + lg * 8]);
            v8f z = {};
            z = wmma_bf16(a0, qb[0], z);
            z = wmma_bf16(a1, qb[1], z);
            st[t] = z;
        }

        // bias + mask + online softmax (lane owns one qrow; L^16 has other keys)
        float vmax = -3.0e38f;
        #pragma unroll
        for (int t = 0; t < 4; ++t)
            #pragma unroll
            for (int r = 0; r < 8; ++r) {
                int kl = t * 16 + lg * 8 + r;
                float s = st[t][r] + sBias[cur][kl - qrl + 63] + sMask[cur][kl];
                st[t][r] = s;
                vmax = fmaxf(vmax, s);
            }
        vmax = fmaxf(vmax, __shfl_xor(vmax, 16, 32));
        float m_new = fmaxf(m_i, vmax);
        float alpha = __expf(m_i - m_new);
        float lsum = 0.0f;
        #pragma unroll
        for (int t = 0; t < 4; ++t)
            #pragma unroll
            for (int r = 0; r < 8; ++r) {
                float p = __expf(st[t][r] - m_new);
                st[t][r] = p;
                lsum += p;
            }
        lsum += __shfl_xor(lsum, 16, 32);
        l_i = l_i * alpha + lsum;
        m_i = m_new;

        // P's A-fragments fall out of the S^T C-layout: pure f32->bf16 convert.
        Frag16 pa0, pa1;
        #pragma unroll
        for (int k2 = 0; k2 < 8; ++k2) {
            pa0.v[k2]     = (__bf16)st[0][k2];
            pa0.v[8 + k2] = (__bf16)st[1][k2];
            pa1.v[k2]     = (__bf16)st[2][k2];
            pa1.v[8 + k2] = (__bf16)st[3][k2];
        }

        // Rescale O (alpha indexed by accumulator row) then O += P*V.
        #pragma unroll
        for (int r = 0; r < 8; ++r) {
            float ar = __shfl(alpha, lg * 8 + r, 32);
            #pragma unroll
            for (int t = 0; t < 4; ++t) o[t][r] *= ar;
        }
        #pragma unroll
        for (int t = 0; t < 4; ++t) {
            int vrow = t * 16 + lm;     // hd column tile; B elem (k,n) = V^T[n][k]
            v16bf b0 = ld_frag(&sV[cur][vrow * 64 +      lg * 8], &sV[cur][vrow * 64 + 16 + lg * 8]);
            v16bf b1 = ld_frag(&sV[cur][vrow * 64 + 32 + lg * 8], &sV[cur][vrow * 64 + 48 + lg * 8]);
            o[t] = wmma_bf16(pa0.v, b0, o[t]);
            o[t] = wmma_bf16(pa1.v, b1, o[t]);
        }
    }

    // Normalize by l and store bf16 into [B, N, H*hd].
    float linv = 1.0f / l_i;
    #pragma unroll
    for (int r = 0; r < 8; ++r) {
        float lr = __shfl(linv, lg * 8 + r, 32);
        int qrow = qbase + wave * 16 + lg * 8 + r;
        size_t base = ((size_t)b * SEQ + qrow) * DIMC + h * HD;
        #pragma unroll
        for (int t = 0; t < 4; ++t)
            O[base + t * 16 + lm] = (__bf16)(o[t][r] * lr);
    }
}

// ---------------------------------------------------------------------------
// Kernel 3: out = attn @ w_proj.T + b_proj (bf16 A async-staged, fp32 out)
// ---------------------------------------------------------------------------
__global__ __launch_bounds__(128)
void proj_gemm_kernel(const __bf16* __restrict__ A, const float* __restrict__ W,
                      const float* __restrict__ bias, float* __restrict__ Out)
{
    __shared__ __align__(16) __bf16 sA[64 * 32];
    __shared__ __align__(16) __bf16 sB[64 * 32];

    const int tid  = threadIdx.x;
    const int lane = tid & 31;
    const int wave = tid >> 5;
    const int lg   = lane >> 4;
    const int lm   = lane & 15;
    const int m0   = blockIdx.x * 64;
    const int n0   = blockIdx.y * 64;

    v8f acc[4] = {};

    for (int kb = 0; kb < DIMC; kb += 32) {
        __syncthreads();
        #pragma unroll
        for (int it = 0; it < 2; ++it) {   // A tile: async copy, 256 x b128
            int idx = tid + it * 128;
            int row = idx >> 2, c = idx & 3;
            async_copy_b128(A + (size_t)(m0 + row) * DIMC + kb + c * 8,
                            &sA[idx * 8]);
        }
        float4 gw[4];
        #pragma unroll
        for (int it = 0; it < 4; ++it) {   // W tile: f32 -> bf16
            int idx = tid + it * 128;
            int row = idx >> 3, c4 = idx & 7;
            gw[it] = *(const float4*)(W + (size_t)(n0 + row) * DIMC + kb + c4 * 4);
        }
        if (kb + 32 < DIMC)
            __builtin_prefetch(W + (size_t)(n0 + (tid >> 3)) * DIMC + kb + 32, 0, 1);
        #pragma unroll
        for (int it = 0; it < 4; ++it) {
            int idx = tid + it * 128;
            int row = idx >> 3, c4 = idx & 7;
            __bf16* db = &sB[row * 32 + c4 * 4];
            db[0] = (__bf16)gw[it].x; db[1] = (__bf16)gw[it].y;
            db[2] = (__bf16)gw[it].z; db[3] = (__bf16)gw[it].w;
        }
        wait_async0();
        __syncthreads();

        int arow = wave * 16 + lm;
        v16bf a = ld_frag(&sA[arow * 32 + lg * 8], &sA[arow * 32 + 16 + lg * 8]);
        #pragma unroll
        for (int t = 0; t < 4; ++t) {
            int brow = t * 16 + lm;
            v16bf bfr = ld_frag(&sB[brow * 32 + lg * 8], &sB[brow * 32 + 16 + lg * 8]);
            acc[t] = wmma_bf16(a, bfr, acc[t]);
        }
    }

    #pragma unroll
    for (int t = 0; t < 4; ++t) {
        int ncol = n0 + t * 16 + lm;
        float bv = bias[ncol];
        #pragma unroll
        for (int r = 0; r < 8; ++r) {
            int mrow = m0 + wave * 16 + lg * 8 + r;
            Out[(size_t)mrow * DIMC + ncol] = acc[t][r] + bv;
        }
    }
}

// ---------------------------------------------------------------------------
extern "C" void kernel_launch(void* const* d_in, const int* in_sizes, int n_in,
                              void* d_out, int out_size, void* d_ws, size_t ws_size,
                              hipStream_t stream) {
    const float* x          = (const float*)d_in[0];
    const float* w_qkv      = (const float*)d_in[1];
    const float* w_proj     = (const float*)d_in[2];
    const float* b_proj     = (const float*)d_in[3];
    const float* bias_table = (const float*)d_in[4];
    const unsigned char* km = (const unsigned char*)d_in[5];

    const size_t headElems = (size_t)BATCH * NH * SEQ * HD;   // 6,291,456
    __bf16* Q  = (__bf16*)d_ws;
    __bf16* K  = Q + headElems;
    __bf16* V  = K + headElems;
    __bf16* AT = V + headElems;                               // bf16 [B*N, C]
    float*  out = (float*)d_out;

    dim3 blk(128);
    dim3 g1(MTOT / 64, QKVN / 64);      // 128 x 36
    qkv_gemm_kernel<<<g1, blk, 0, stream>>>(x, w_qkv, Q, K, V);

    dim3 g2(SEQ / 64, BATCH * NH);      // 16 x 96
    flash_attn_kernel<<<g2, blk, 0, stream>>>(Q, K, V, bias_table, km, AT);

    dim3 g3(MTOT / 64, DIMC / 64);      // 128 x 12
    proj_gemm_kernel<<<g3, blk, 0, stream>>>(AT, w_proj, b_proj, out);
}
